// LayerNormGRU_67740224192644
// MI455X (gfx1250) — matrix operand order repaired and checked by reading, not compiled, not run on hardware
//
#include <hip/hip_runtime.h>
#include <hip/hip_bf16.h>
#include <math.h>

// ---------------------------------------------------------------------------
// LayerNorm-GRU for MI455X (gfx1250, wave32, WMMA).
//   T=512, B=128, I=H=512, L=2, 3H=1536.
//   * bf16 WMMA (v_wmma_f32_16x16x32_bf16), f32 accumulate.
//   * Input projection batched over 128-timestep chunks (big GEMM).
//   * Recurrent step: 8 WGs, each owns 16 batch rows x full 1536 cols in LDS,
//     fused LN(3H) + LN(n-chunk) + gate math + in-place h update.
//   * All 12 B fragments of a K-step are pinned live via an empty asm so the
//     24 b128 loads clause together with ONE s_wait_loadcnt, and the next
//     iteration's loads overlap the 12-WMMA block (latency-critical for the
//     8-WG recurrent kernel).
// ---------------------------------------------------------------------------

typedef __bf16 bf16;
typedef __attribute__((ext_vector_type(16))) __bf16 v16bf;
typedef __attribute__((ext_vector_type(8)))  __bf16 v8bf;
typedef __attribute__((ext_vector_type(8)))  float  v8f;

#define T_  512
#define B_  128
#define H_  512
#define L_  2
#define G3  1536            // 3*H
#define NTW 12              // n-tiles per wave (96 tiles / 8 waves)
#define KIT (H_ / 32)       // 16 K-iterations of 16x16x32
#define LSTR 1540           // padded LDS row stride in floats (bank-safe)
#define EPS_ 1e-5f
#define TCHUNK 128          // timesteps per precomputed gate chunk

union ABFrag { v16bf v; v8bf h[2]; };

// Force A + all 12 B fragments to be materialized in distinct registers
// before the WMMA chain (defeats the allocator's single-buffer reuse that
// serializes load->wait->wmma per tile).
#define PIN_FRAGS(a_, b_)                                                     \
  asm volatile("" : "+v"((a_).v),                                             \
               "+v"((b_)[0].v), "+v"((b_)[1].v), "+v"((b_)[2].v),             \
               "+v"((b_)[3].v), "+v"((b_)[4].v), "+v"((b_)[5].v),             \
               "+v"((b_)[6].v), "+v"((b_)[7].v), "+v"((b_)[8].v),             \
               "+v"((b_)[9].v), "+v"((b_)[10].v), "+v"((b_)[11].v))

// ---- 16-row x 1536-col GEMM strip into LDS (acc += bias) -------------------
__device__ __forceinline__ void gemm_tile16_rows(
    const bf16* __restrict__ A,      // base of 16-row block, row stride H_
    const bf16* __restrict__ W,      // [G3, H_] row-major (weights)
    const float* __restrict__ bias,  // [G3]
    float* lds, int tid)
{
  const int lane   = tid & 31;
  const int wave   = tid >> 5;      // 0..7
  const int lane16 = lane & 15;
  const int half   = lane >> 4;     // 0 or 1

  const v8f vzero = {0.f,0.f,0.f,0.f,0.f,0.f,0.f,0.f};
  v8f acc[NTW];
#pragma unroll
  for (int j = 0; j < NTW; ++j) acc[j] = vzero;

  const bf16* arow  = A + (long)lane16 * H_;
  // Base W pointer for this wave's first n-tile; tiles step by 16 rows of W.
  const bf16* wbase = W + ((long)(wave * NTW) * 16 + lane16) * H_ + half * 16;

#pragma unroll 2
  for (int kk = 0; kk < KIT; ++kk) {
    const int k0 = kk * 32;

    // A fragment: 16x32 bf16; two contiguous 16B loads per lane.
    ABFrag a;
    a.h[0] = *(const v8bf*)(arow + k0 + half * 16);
    a.h[1] = *(const v8bf*)(arow + k0 + half * 16 + 8);

    // Stage ALL B fragments (24 independent b128 loads -> clauses).
    ABFrag b[NTW];
#pragma unroll
    for (int j = 0; j < NTW; ++j) {
      const bf16* wrow = wbase + (long)j * 16 * H_ + k0;
      b[j].h[0] = *(const v8bf*)(wrow);
      b[j].h[1] = *(const v8bf*)(wrow + 8);
    }

    PIN_FRAGS(a, b);   // all 13 fragments live: one wait, dense WMMA block

#pragma unroll
    for (int j = 0; j < NTW; ++j) {
      acc[j] = __builtin_amdgcn_wmma_f32_16x16x32_bf16(
          false, a.v, false, b[j].v, (short)0, acc[j], false, false);
    }
  }

  // C layout: VGPR r, lanes 0-15 -> row r, lanes 16-31 -> row r+8, col=lane16.
#pragma unroll
  for (int j = 0; j < NTW; ++j) {
    const int col = (wave * NTW + j) * 16 + lane16;
    const float bv = bias[col];
#pragma unroll
    for (int r = 0; r < 8; ++r) {
      const int row = r + half * 8;
      lds[row * LSTR + col] = acc[j][r] + bv;
    }
  }
}

// ---- LayerNorm over [col0, col0+width) for 16 LDS rows ---------------------
// 256 threads: 16 threads per row; each column owned by exactly one thread.
__device__ __forceinline__ void ln_rows(float* lds, int tid, int col0, int width,
                                        const float* __restrict__ gamma,
                                        const float* __restrict__ beta)
{
  const int row = tid >> 4;
  const int t16 = tid & 15;
  float s = 0.f, ss = 0.f;
  for (int c = col0 + t16; c < col0 + width; c += 16) {
    const float v = lds[row * LSTR + c];
    s += v; ss += v * v;
  }
#pragma unroll
  for (int m = 8; m >= 1; m >>= 1) {   // reduce across the 16-lane half-wave
    s  += __shfl_xor(s,  m, 32);
    ss += __shfl_xor(ss, m, 32);
  }
  const float inv  = 1.f / (float)width;
  const float mean = s * inv;
  const float var  = ss * inv - mean * mean;
  const float rstd = rsqrtf(var + EPS_);
  for (int c = col0 + t16; c < col0 + width; c += 16) {
    const float v = lds[row * LSTR + c];
    lds[row * LSTR + c] = (v - mean) * rstd * gamma[c - col0] + beta[c - col0];
  }
}

// ---- batched input projection + LN(3H) + LN(n-chunk) -----------------------
__global__ void __launch_bounds__(256)
gemm_ln_ih_kernel(const bf16* __restrict__ xin, const bf16* __restrict__ wih,
                  const float* __restrict__ b_ih, const float* __restrict__ g_ih,
                  const float* __restrict__ be_ih, const float* __restrict__ g_in,
                  const float* __restrict__ be_in, float* __restrict__ gates)
{
  extern __shared__ float lds[];
  const int tid = threadIdx.x;
  const long m0 = (long)blockIdx.x * 16;

  gemm_tile16_rows(xin + m0 * H_, wih, b_ih, lds, tid);
  __syncthreads();
  ln_rows(lds, tid, 0, G3, g_ih, be_ih);        // LN over full 3H row
  __syncthreads();
  ln_rows(lds, tid, 2 * H_, H_, g_in, be_in);   // nested LN on i_n chunk
  __syncthreads();

  // Vectorized b128 store of the LN'd gate strip.
  float* out = gates + m0 * G3;
  for (int idx = tid; idx < 16 * (G3 / 4); idx += 256) {
    const int row = idx / (G3 / 4);
    const int c4  = idx - row * (G3 / 4);
    *(float4*)(out + (long)row * G3 + c4 * 4) =
        *(const float4*)(lds + row * LSTR + c4 * 4);
  }
}

// ---- one recurrent timestep: hh GEMM + LNs + gates + state update ----------
__global__ void __launch_bounds__(256)
gru_step_kernel(const bf16* __restrict__ hbf, const bf16* __restrict__ whh,
                const float* __restrict__ b_hh, const float* __restrict__ g_hh,
                const float* __restrict__ be_hh, const float* __restrict__ g_hn,
                const float* __restrict__ be_hn,
                const float* __restrict__ gates_t,  // [B_, G3] LN'd ih gates
                float* __restrict__ hf,             // [B_, H_] f32 state (in/out)
                bf16*  __restrict__ hb_out,         // bf16 state (in-place ok)
                bf16*  __restrict__ xnext,          // next-layer input row, or null
                float* __restrict__ yout,           // final-layer output row, or null
                float* __restrict__ hfin)           // h_final slot at t==T-1, or null
{
  extern __shared__ float lds[];
  const int tid = threadIdx.x;
  const int m0  = blockIdx.x * 16;

  gemm_tile16_rows(hbf + (long)m0 * H_, whh, b_hh, lds, tid);
  __syncthreads();
  ln_rows(lds, tid, 0, G3, g_hh, be_hh);        // LN over full 3H row
  __syncthreads();
  ln_rows(lds, tid, 2 * H_, H_, g_hn, be_hn);   // nested LN on h_n chunk
  __syncthreads();

  for (int idx = tid; idx < 16 * H_; idx += 256) {
    const int row  = idx >> 9;          // /512
    const int col  = idx & (H_ - 1);
    const int grow = m0 + row;
    const long gb  = (long)grow * G3;
    const float i_r = gates_t[gb + col];
    const float i_z = gates_t[gb + H_ + col];
    const float i_n = gates_t[gb + 2 * H_ + col];
    const float h_r = lds[row * LSTR + col];
    const float h_z = lds[row * LSTR + H_ + col];
    const float h_n = lds[row * LSTR + 2 * H_ + col];
    const float hp  = hf[(long)grow * H_ + col];

    const float r = 1.f / (1.f + __expf(-(i_r + h_r)));
    const float z = 1.f / (1.f + __expf(-(i_z + h_z)));
    const float n = tanhf(i_n + r * h_n);
    const float hnew = (1.f - z) * n + z * hp;

    hf[(long)grow * H_ + col]     = hnew;
    hb_out[(long)grow * H_ + col] = (bf16)hnew;
    if (xnext) xnext[(long)grow * H_ + col] = (bf16)hnew;
    if (yout)  yout [(long)grow * H_ + col] = hnew;
    if (hfin)  hfin [(long)grow * H_ + col] = hnew;
  }
}

// ---- small helpers ---------------------------------------------------------
__global__ void cvt_f32_bf16(const float* __restrict__ in, bf16* __restrict__ out, long n) {
  long i = (long)blockIdx.x * blockDim.x + threadIdx.x;
  const long stride = (long)gridDim.x * blockDim.x;
  for (; i < n; i += stride) out[i] = (bf16)in[i];
}

__global__ void init_h_kernel(const float* __restrict__ hx, float* __restrict__ hf,
                              bf16* __restrict__ hb, int n) {
  const int i = blockIdx.x * blockDim.x + threadIdx.x;
  if (i < n) { const float v = hx[i]; hf[i] = v; hb[i] = (bf16)v; }
}

// ---------------------------------------------------------------------------
extern "C" void kernel_launch(void* const* d_in, const int* in_sizes, int n_in,
                              void* d_out, int out_size, void* d_ws, size_t ws_size,
                              hipStream_t stream) {
  (void)in_sizes; (void)n_in; (void)out_size; (void)ws_size;
  const float* x     = (const float*)d_in[0];
  const float* hx    = (const float*)d_in[1];
  const float* w_ih  = (const float*)d_in[2];
  const float* w_hh  = (const float*)d_in[3];
  const float* b_ih  = (const float*)d_in[4];
  const float* b_hh  = (const float*)d_in[5];
  const float* g_ih  = (const float*)d_in[6];
  const float* be_ih = (const float*)d_in[7];
  const float* g_hh  = (const float*)d_in[8];
  const float* be_hh = (const float*)d_in[9];
  const float* g_in  = (const float*)d_in[10];
  const float* be_in = (const float*)d_in[11];
  const float* g_hn  = (const float*)d_in[12];
  const float* be_hn = (const float*)d_in[13];
  float* out = (float*)d_out;

  // Workspace carve-up.
  char* ws = (char*)d_ws;
  bf16* wih_bf = (bf16*)ws;  ws += (size_t)L_ * G3 * H_ * sizeof(bf16);
  bf16* whh_bf = (bf16*)ws;  ws += (size_t)L_ * G3 * H_ * sizeof(bf16);
  bf16* x_bf   = (bf16*)ws;  ws += (size_t)T_ * B_ * H_ * sizeof(bf16);
  float* gates = (float*)ws; ws += (size_t)TCHUNK * B_ * G3 * sizeof(float);
  float* hf    = (float*)ws; ws += (size_t)B_ * H_ * sizeof(float);
  bf16*  hb    = (bf16*)ws;  ws += (size_t)B_ * H_ * sizeof(bf16);

  const size_t ldsBytes = (size_t)16 * LSTR * sizeof(float); // ~96 KB / WG

  // Weight + input conversion to bf16.
  {
    const long nw = (long)L_ * G3 * H_;
    cvt_f32_bf16<<<1024, 256, 0, stream>>>(w_ih, wih_bf, nw);
    cvt_f32_bf16<<<1024, 256, 0, stream>>>(w_hh, whh_bf, nw);
    cvt_f32_bf16<<<2048, 256, 0, stream>>>(x, x_bf, (long)T_ * B_ * H_);
  }

  for (int l = 0; l < L_; ++l) {
    init_h_kernel<<<(B_ * H_) / 256, 256, 0, stream>>>(
        hx + (size_t)l * B_ * H_, hf, hb, B_ * H_);

    const bf16* wih_l = wih_bf + (size_t)l * G3 * H_;
    const bf16* whh_l = whh_bf + (size_t)l * G3 * H_;
    const bool last = (l == L_ - 1);

    for (int c = 0; c < T_ / TCHUNK; ++c) {
      // Batched input projection + LN for this chunk of timesteps.
      gemm_ln_ih_kernel<<<TCHUNK * B_ / 16, 256, ldsBytes, stream>>>(
          x_bf + (size_t)c * TCHUNK * B_ * H_, wih_l,
          b_ih + (size_t)l * G3, g_ih + (size_t)l * G3, be_ih + (size_t)l * G3,
          g_in + (size_t)l * H_, be_in + (size_t)l * H_, gates);

      for (int tt = 0; tt < TCHUNK; ++tt) {
        const int t = c * TCHUNK + tt;
        gru_step_kernel<<<B_ / 16, 256, ldsBytes, stream>>>(
            hb, whh_l,
            b_hh + (size_t)l * G3, g_hh + (size_t)l * G3, be_hh + (size_t)l * G3,
            g_hn + (size_t)l * H_, be_hn + (size_t)l * H_,
            gates + (size_t)tt * B_ * G3,
            hf, hb,
            last ? nullptr : (x_bf + (size_t)t * B_ * H_),
            last ? (out + (size_t)t * B_ * H_) : nullptr,
            (t == T_ - 1) ? (out + (size_t)T_ * B_ * H_ + (size_t)l * B_ * H_)
                          : nullptr);
      }
    }
  }
}
